// slae_1_46729244181195
// MI455X (gfx1250) — matrix operand order repaired and checked by reading, not compiled, not run on hardware
//
#include <hip/hip_runtime.h>

typedef __attribute__((ext_vector_type(2)))  float    v2f;
typedef __attribute__((ext_vector_type(8)))  float    v8f;
typedef __attribute__((ext_vector_type(8)))  _Float16 v8h;
typedef __attribute__((ext_vector_type(16))) _Float16 v16h;

#define Bc 32
#define Cc 3
#define Tc 25
#define Hc 64
#define Dc 256
#define MROWS   (Bc*Cc*Hc)          /* 6144 rows per timestep             */
#define OUTHALF (Bc*Cc*Tc*Hc*Dc)    /* 39,321,600 elems per output tensor */

// Pre-convert Wd (256x256 f32) to f16 for the spike-side GEMM (spk is exactly
// representable in f16, so only the weights are quantized: ~5e-4 rel).
__global__ void wd_to_f16_kernel(const float* __restrict__ Wd,
                                 _Float16* __restrict__ Wdh) {
    int i = blockIdx.x * 256 + threadIdx.x;   // 256 blocks x 256 thr = 65536
    Wdh[i] = (_Float16)Wd[i];
}

// Low 32 bits of an LDS pointer == LDS byte offset (LDS aperture has the
// offset in addr[31:0]; hardware truncates per the flat-addressing rule).
static __device__ __forceinline__ unsigned lds_lo32(const void* p) {
    return (unsigned)(uintptr_t)p;
}

// Issue one 16-byte async global->LDS transfer per lane for a 4 KB chunk.
// Per ISA: one immediate offset applies to both the LDS and the global address.
#define ASYNC_TILE_CHUNK(ldsa, voff, sbase, imm)                              \
    asm volatile("global_load_async_to_lds_b128 %0, %1, %2 offset:" #imm      \
                 :: "v"(ldsa), "v"(voff), "s"(sbase) : "memory")

static __device__ __forceinline__
void issue_x_tile_async(const float* __restrict__ gbase, unsigned lds_base,
                        int tid) {
    unsigned ldsa = lds_base + tid * 16;   // per-lane LDS dest (b128)
    unsigned voff = tid * 16;              // per-lane global byte offset
    ASYNC_TILE_CHUNK(ldsa, voff, gbase, 0);
    ASYNC_TILE_CHUNK(ldsa, voff, gbase, 4096);
    ASYNC_TILE_CHUNK(ldsa, voff, gbase, 8192);
    ASYNC_TILE_CHUNK(ldsa, voff, gbase, 12288);
}

__global__ __launch_bounds__(256)
void snn_fused_kernel(const float* __restrict__ x,
                      const float* __restrict__ We,
                      const float* __restrict__ be,
                      const float* __restrict__ bd,
                      const float* __restrict__ beta,
                      const _Float16* __restrict__ Wdh,
                      float* __restrict__ out)
{
    __shared__ __align__(16) float    xs[2][16 * Dc]; // 32 KB: double-buffered x tile
    __shared__ __align__(16) _Float16 sp[16 * Dc];    //  8 KB: spike tile (f16)

    const int tid  = threadIdx.x;
    const int lane = tid & 31;          // wave32
    const int wave = tid >> 5;          // 8 waves
    const int hi   = lane >> 4;         // 0 or 1 (half-wave)
    const int l15  = lane & 15;

    // 16-row tile of the 6144 (b,c,h) rows; 16 | 64 so tile never crosses h-group
    const int row0   = blockIdx.x * 16;
    const int rowgrp = row0 / Hc;       // flattened (b*C + c)
    const int hbase  = row0 % Hc;

    // each wave owns 32 columns = two 16-col WMMA N-tiles
    const int n0[2] = { wave * 32, wave * 32 + 16 };

    float bev[2], bdv[2], btv[2];
    for (int j = 0; j < 2; ++j) {
        int e  = n0[j] + l15;           // column of every C/D element this lane holds
        bev[j] = be[e];
        bdv[j] = bd[e];
        btv[j] = fminf(fmaxf(beta[e], 0.0f), 1.0f);
    }

    // persistent membrane state: 2 tiles x 8 VGPR-rows per lane
    float mem[2][8];
    for (int j = 0; j < 2; ++j)
        for (int r = 0; r < 8; ++r) mem[j][r] = 0.0f;

    const int am = l15;                 // A-fragment row (both GEMMs)
    const int ak = hi << 1;             // f32 A/B K sub-offset: 0 or 2

    // [B,C,T,H,D] tile base for timestep t (identical layout for x/spk/dec)
    const long tstride = (long)Hc * Dc;                       // t -> t+1
    const long ob0 = ((long)rowgrp * Tc * Hc + hbase) * Dc;   // t = 0

    // preload x tile for t=0 into buffer 0
    issue_x_tile_async(x + ob0, lds_lo32(&xs[0][0]), tid);

    for (int t = 0; t < Tc; ++t) {
        const int p = t & 1;
        const long ob = ob0 + (long)t * tstride;
        const float* __restrict__ xsb = &xs[p][0];

        asm volatile("s_wait_asynccnt 0x0" ::: "memory");  // own async landed
        __syncthreads();                                   // everyone's landed

        // ---- GEMM1: enc = x_t @ We^T + be   (V_WMMA_F32_16X16X4_F32, exact fp32)
        v8f acc[2];
        for (int j = 0; j < 2; ++j)
            for (int r = 0; r < 8; ++r) acc[j][r] = bev[j];   // fold bias into C

        for (int k0 = 0; k0 < Dc; k0 += 4) {
            v2f a = *(const v2f*)&xsb[am * Dc + k0 + ak];      // ds_load_b64
            for (int j = 0; j < 2; ++j) {
                // B[k,n] = We[n,k]  (enc = X @ We^T); L2-resident weight read
                v2f b = *(const v2f*)&We[(n0[j] + l15) * Dc + k0 + ak];
                acc[j] = __builtin_amdgcn_wmma_f32_16x16x4_f32(
                    false, a, false, b, (short)0, acc[j], false, false);
            }
        }

        // ---- elementwise LIF scan step (state lives in VGPRs across t) ----
        float* __restrict__ outs = out + ob;            // uniform base -> SADDR
        for (int j = 0; j < 2; ++j) {
            for (int r = 0; r < 8; ++r) {
                float m_old = mem[j][r];
                float reset = (m_old > 1.0f) ? 1.0f : 0.0f;   // heaviside(mem-1)
                float m_new = btv[j] * m_old + acc[j][r] - reset;
                float s     = (m_new > 1.0f) ? 1.0f : 0.0f;   // spike
                mem[j][r] = m_new;
                int lm  = r + (hi << 3);                       // local row 0..15
                int idx = lm * Dc + n0[j] + l15;
                outs[idx] = s;                                 // spk_rec (f32)
                sp[idx]   = (_Float16)s;                       // exact in f16
            }
        }
        __syncthreads();   // spike tile visible to all waves (K spans all cols)

        // prefetch next timestep's x tile while GEMM2 runs (async -> LDS)
        if (t + 1 < Tc)
            issue_x_tile_async(x + ob + tstride, lds_lo32(&xs[p ^ 1][0]), tid);

        // ---- GEMM2: dec = sigmoid(spk @ Wd^T + bd)  (V_WMMA_F32_16X16X32_F16)
        v8f dacc[2];
        for (int j = 0; j < 2; ++j)
            for (int r = 0; r < 8; ++r) dacc[j][r] = bdv[j];

        for (int k0 = 0; k0 < Dc; k0 += 32) {
            // A 16x32 in this code's contiguous K order: lanes0-15 K=[k0,k0+16),
            // lanes16-31 K=[k0+16,k0+32) (matches B).
            int kb = k0 + (hi << 4);
            v8h alo = *(const v8h*)&sp[am * Dc + kb];          // ds_load_b128
            v8h ahi = *(const v8h*)&sp[am * Dc + kb + 8];
            v16h aa;
            for (int i2 = 0; i2 < 8; ++i2) { aa[i2] = alo[i2]; aa[i2 + 8] = ahi[i2]; }
            for (int j = 0; j < 2; ++j) {
                // 16-bit B 32x16: 16 consecutive halves at K = k0 + 16*hi
                int n = n0[j] + l15;
                v16h bb = *(const v16h*)&Wdh[n * Dc + k0 + (hi << 4)];
                dacc[j] = __builtin_amdgcn_wmma_f32_16x16x32_f16(
                    false, aa, false, bb, (short)0, dacc[j], false, false);
            }
        }

        float* __restrict__ outd = out + (OUTHALF + ob);
        for (int j = 0; j < 2; ++j) {
            for (int r = 0; r < 8; ++r) {
                float v   = dacc[j][r];
                float dsg = 1.0f / (1.0f + __expf(-v));
                int lm  = r + (hi << 3);
                int idx = lm * Dc + n0[j] + l15;
                outd[idx] = dsg;                               // dec_rec
            }
        }
        __syncthreads();   // protect sp before next iteration's scan writes
    }
}

extern "C" void kernel_launch(void* const* d_in, const int* in_sizes, int n_in,
                              void* d_out, int out_size, void* d_ws, size_t ws_size,
                              hipStream_t stream) {
    const float* x    = (const float*)d_in[0];
    const float* We   = (const float*)d_in[1];
    const float* be   = (const float*)d_in[2];
    const float* Wd   = (const float*)d_in[3];
    const float* bd   = (const float*)d_in[4];
    const float* beta = (const float*)d_in[5];
    float* out = (float*)d_out;
    _Float16* Wdh = (_Float16*)d_ws;    // 128 KB scratch for f16 weights

    wd_to_f16_kernel<<<256, 256, 0, stream>>>(Wd, Wdh);
    snn_fused_kernel<<<MROWS / 16, 256, 0, stream>>>(x, We, be, bd, beta, Wdh, out);
}
